// LSTMLayer_78460462563575
// MI455X (gfx1250) — matrix operand order repaired and checked by reading, not compiled, not run on hardware
//
#include <hip/hip_runtime.h>
#include <hip/hip_bf16.h>

typedef __bf16 bhalf_t;
typedef __attribute__((ext_vector_type(16))) __bf16 v16bf;
typedef __attribute__((ext_vector_type(8)))  __bf16 v8bf;
typedef __attribute__((ext_vector_type(8)))  float  v8f;

#define SEQ   512
#define BATCH 64
#define ISZ   1024
#define HSZ   1024
#define KWF   (ISZ + 2 * HSZ)   /* 3072 : K for write/forget/output gates      */
#define KCA   (ISZ + HSZ)       /* 2048 : K for candidate (x,h prefix of comb) */

// ---------------------------------------------------------------------------
// Fragment loaders using contiguous k order, the same for A and B
// A 16x32 bf16: lane group g = lane>>4 holds K = g*16+{0..15} of row m = lane&15
// B 32x16 bf16: lane group g holds K = g*16+{0..15} of column n = lane&15
// ---------------------------------------------------------------------------
__device__ __forceinline__ v16bf load_frag_a(const bhalf_t* __restrict__ row, int koff) {
  v16bf a;
  v8bf* ap = reinterpret_cast<v8bf*>(&a);
  ap[0] = *reinterpret_cast<const v8bf*>(row + koff);
  ap[1] = *reinterpret_cast<const v8bf*>(row + koff + 8);
  return a;
}

__device__ __forceinline__ v16bf load_frag_b(const bhalf_t* __restrict__ wrow, int koff) {
  v16bf b;
  v8bf* bp = reinterpret_cast<v8bf*>(&b);
  bp[0] = *reinterpret_cast<const v8bf*>(wrow + koff);
  bp[1] = *reinterpret_cast<const v8bf*>(wrow + koff + 8);
  return b;
}

__device__ __forceinline__ v8f wmma_bf16(v16bf a, v16bf b, v8f c) {
  return __builtin_amdgcn_wmma_f32_16x16x32_bf16(false, a, false, b, (short)0, c,
                                                 false, false);
}

__device__ __forceinline__ float sigmoidf_fast(float x) {
  return 1.0f / (1.0f + __expf(-x));
}

// ---------------------------------------------------------------------------
// Phase A (per step): w = sig(comb Ww^T + bw), f = sig(comb Wf^T + bf),
// g = tanh(comb[:,0:2048] Wc^T + bc), c_new = f*c + w*g.
// comb = [x_t | h | c] bf16, [64, 3072] row-major (ping-pong pair).
// One wave owns one 16x16 tile of all three gates; software-pipelined K loop
// (next chunk's loads issued before current chunk's WMMAs).
// grid = (64 N-tiles, 4 M-tiles), block = 32 threads (one wave).
// ---------------------------------------------------------------------------
__global__ __launch_bounds__(32)
void lstm_phase_a(const bhalf_t* __restrict__ comb,      // [64,3072] current
                  bhalf_t*       __restrict__ comb_nxt,  // [64,3072] next (c cols written)
                  float*         __restrict__ cfp,       // [64,1024] f32 cell state
                  const bhalf_t* __restrict__ Ww,        // [1024,3072] bf16 row-major
                  const bhalf_t* __restrict__ Wf,        // [1024,3072]
                  const bhalf_t* __restrict__ Wc,        // [1024,2048]
                  const float*   __restrict__ bw,
                  const float*   __restrict__ bfg,
                  const float*   __restrict__ bc,
                  float*         __restrict__ c_tail)    // d_out c_f slot or null
{
  const int lane  = threadIdx.x & 31;
  const int g     = lane >> 4;
  const int mbase = blockIdx.y * 16;
  const int nbase = blockIdx.x * 16;

  const int arow  = mbase + (lane & 15);
  const int akoff = g * 16;
  const int nb    = nbase + (lane & 15);
  const int bkoff = g * 16;

  const bhalf_t* arp   = comb + (size_t)arow * KWF;
  const bhalf_t* wwrow = Ww + (size_t)nb * KWF;
  const bhalf_t* wfrow = Wf + (size_t)nb * KWF;
  const bhalf_t* wcrow = Wc + (size_t)nb * KCA;

  v8f accw = {}, accf = {}, accc = {};

  // --- software-pipelined K loop: prologue (chunk 0) ---
  v16bf a  = load_frag_a(arp,   akoff);
  v16bf vw = load_frag_b(wwrow, bkoff);
  v16bf vf = load_frag_b(wfrow, bkoff);
  v16bf vc = load_frag_b(wcrow, bkoff);

  // chunks 0..62 consumed (all feed the candidate); chunk k loaded ahead
#pragma unroll 2
  for (int k = 32; k < KCA; k += 32) {
    v16bf a_n  = load_frag_a(arp + k,   akoff);
    v16bf vw_n = load_frag_b(wwrow + k, bkoff);
    v16bf vf_n = load_frag_b(wfrow + k, bkoff);
    v16bf vc_n = load_frag_b(wcrow + k, bkoff);
    accw = wmma_bf16(a, vw, accw);
    accf = wmma_bf16(a, vf, accf);
    accc = wmma_bf16(a, vc, accc);
    a = a_n; vw = vw_n; vf = vf_n; vc = vc_n;
  }
  // consume chunk 63 (last with candidate); preload chunk 64 (w/f only)
  {
    v16bf a_n  = load_frag_a(arp + KCA,   akoff);
    v16bf vw_n = load_frag_b(wwrow + KCA, bkoff);
    v16bf vf_n = load_frag_b(wfrow + KCA, bkoff);
    accw = wmma_bf16(a, vw, accw);
    accf = wmma_bf16(a, vf, accf);
    accc = wmma_bf16(a, vc, accc);
    a = a_n; vw = vw_n; vf = vf_n;
  }
  // chunks 64..94 consumed (w/f only)
#pragma unroll 2
  for (int k = KCA + 32; k < KWF; k += 32) {
    v16bf a_n  = load_frag_a(arp + k,   akoff);
    v16bf vw_n = load_frag_b(wwrow + k, bkoff);
    v16bf vf_n = load_frag_b(wfrow + k, bkoff);
    accw = wmma_bf16(a, vw, accw);
    accf = wmma_bf16(a, vf, accf);
    a = a_n; vw = vw_n; vf = vf_n;
  }
  // epilogue: chunk 95
  accw = wmma_bf16(a, vw, accw);
  accf = wmma_bf16(a, vf, accf);

  const float biasw = bw[nb], biasf = bfg[nb], biasc = bc[nb];

  // D layout: VGPR v holds M = mbase + v + 8*(lane>>4), N = nb
#pragma unroll
  for (int v = 0; v < 8; ++v) {
    const int brow = mbase + v + g * 8;
    const int sidx = brow * HSZ + nb;
    const float wv = sigmoidf_fast(accw[v] + biasw);
    const float fv = sigmoidf_fast(accf[v] + biasf);
    const float gv = tanhf(accc[v] + biasc);
    const float cn = fv * cfp[sidx] + wv * gv;
    cfp[sidx] = cn;
    comb_nxt[(size_t)brow * KWF + ISZ + HSZ + nb] = (bhalf_t)cn;  // c column slot
    if (c_tail) c_tail[sidx] = cn;
  }
}

// ---------------------------------------------------------------------------
// Phase B (per step): o = sig([x,h,c_new] Wo^T + bo), h_new = o * tanh(c_new).
// Reads x,h from comb (cur) and c_new from comb_nxt; writes outputs[t] (f32),
// h_new bf16 into comb_nxt, and prefills comb_nxt's x columns with x_{t+1}.
// ---------------------------------------------------------------------------
__global__ __launch_bounds__(32)
void lstm_phase_b(const bhalf_t* __restrict__ comb,      // current (x,h cols)
                  bhalf_t*       __restrict__ comb_nxt,  // c_new cols read; h,x cols written
                  const float*   __restrict__ cfp,
                  const bhalf_t* __restrict__ Wo,        // [1024,3072]
                  const float*   __restrict__ bo,
                  const float*   __restrict__ xnext,     // fp32 x_{t+1} slice or null
                  float*         __restrict__ hout,      // d_out + t*BATCH*HSZ
                  float*         __restrict__ h_tail)    // d_out h_f slot or null
{
  const int lane  = threadIdx.x & 31;
  const int g     = lane >> 4;
  const int mbase = blockIdx.y * 16;
  const int nbase = blockIdx.x * 16;

  const int arow  = mbase + (lane & 15);
  const int akoff = g * 16;
  const int nb    = nbase + (lane & 15);
  const int bkoff = g * 16;

  const bhalf_t* arp0  = comb     + (size_t)arow * KWF;  // k in [0, 2048): x,h
  const bhalf_t* arp1  = comb_nxt + (size_t)arow * KWF;  // k in [2048, 3072): c_new
  const bhalf_t* worow = Wo + (size_t)nb * KWF;

  v8f acco = {};

  v16bf a  = load_frag_a(arp0,  akoff);
  v16bf vo = load_frag_b(worow, bkoff);

#pragma unroll 2
  for (int k = 32; k < KWF; k += 32) {
    const bhalf_t* base = (k < KCA) ? arp0 : arp1;   // uniform select
    v16bf a_n  = load_frag_a(base + k,  akoff);
    v16bf vo_n = load_frag_b(worow + k, bkoff);
    acco = wmma_bf16(a, vo, acco);
    a = a_n; vo = vo_n;
  }
  acco = wmma_bf16(a, vo, acco);

  const float biaso = bo[nb];

#pragma unroll
  for (int v = 0; v < 8; ++v) {
    const int brow = mbase + v + g * 8;
    const int sidx = brow * HSZ + nb;
    const float ov = sigmoidf_fast(acco[v] + biaso);
    const float hn = ov * tanhf(cfp[sidx]);
    hout[sidx] = hn;
    comb_nxt[(size_t)brow * KWF + ISZ + nb] = (bhalf_t)hn;          // h column slot
    if (xnext) comb_nxt[(size_t)brow * KWF + nb] = (bhalf_t)xnext[sidx]; // x prefill
    if (h_tail) h_tail[sidx] = hn;
  }
}

// ---------------------------------------------------------------------------
// One-time pass: weights fp32 -> bf16; build comb_0 = [x_0 | h0 | c0] bf16
// and the f32 cell state. After this the recurrent working set (~23 MB) is
// fully L2-resident on the 192 MB L2.
// ---------------------------------------------------------------------------
__global__ void cvt_f32_to_bf16(const float* __restrict__ src,
                                bhalf_t* __restrict__ dst, int n) {
  int i = blockIdx.x * blockDim.x + threadIdx.x;
  const int stride = gridDim.x * blockDim.x;
  for (; i < n; i += stride) dst[i] = (bhalf_t)src[i];
}

__global__ void init_comb(const float* __restrict__ x0,
                          const float* __restrict__ h0,
                          const float* __restrict__ c0,
                          bhalf_t* __restrict__ comb0,
                          float* __restrict__ cfp, int n) {
  int i = blockIdx.x * blockDim.x + threadIdx.x;
  const int stride = gridDim.x * blockDim.x;
  for (; i < n; i += stride) {
    const int b = i >> 10, col = i & 1023;
    const size_t base = (size_t)b * KWF;
    comb0[base + col]             = (bhalf_t)x0[i];
    comb0[base + ISZ + col]       = (bhalf_t)h0[i];
    comb0[base + ISZ + HSZ + col] = (bhalf_t)c0[i];
    cfp[i] = c0[i];
  }
}

extern "C" void kernel_launch(void* const* d_in, const int* in_sizes, int n_in,
                              void* d_out, int out_size, void* d_ws, size_t ws_size,
                              hipStream_t stream) {
  const float* x   = (const float*)d_in[0];
  const float* h0  = (const float*)d_in[1];
  const float* c0  = (const float*)d_in[2];
  const float* Ww  = (const float*)d_in[3];
  const float* bw  = (const float*)d_in[4];
  const float* Wf  = (const float*)d_in[5];
  const float* bfg = (const float*)d_in[6];
  const float* Wc  = (const float*)d_in[7];
  const float* bc  = (const float*)d_in[8];
  const float* Wo  = (const float*)d_in[9];
  const float* bo  = (const float*)d_in[10];
  float* out = (float*)d_out;

  // Workspace carve-out (256B aligned slices), ~23.5 MB total
  char* p = (char*)d_ws;
  auto take = [&p](size_t bytes) -> char* {
    char* r = p;
    p += (bytes + 255) & ~(size_t)255;
    return r;
  };
  bhalf_t* Wwb   = (bhalf_t*)take((size_t)HSZ * KWF * 2);
  bhalf_t* Wfb   = (bhalf_t*)take((size_t)HSZ * KWF * 2);
  bhalf_t* Wcb   = (bhalf_t*)take((size_t)HSZ * KCA * 2);
  bhalf_t* Wob   = (bhalf_t*)take((size_t)HSZ * KWF * 2);
  bhalf_t* comb0 = (bhalf_t*)take((size_t)BATCH * KWF * 2);
  bhalf_t* comb1 = (bhalf_t*)take((size_t)BATCH * KWF * 2);
  float*   cfp   = (float*)  take((size_t)BATCH * HSZ * 4);
  bhalf_t* cmb[2] = {comb0, comb1};

  cvt_f32_to_bf16<<<1024, 256, 0, stream>>>(Ww, Wwb, HSZ * KWF);
  cvt_f32_to_bf16<<<1024, 256, 0, stream>>>(Wf, Wfb, HSZ * KWF);
  cvt_f32_to_bf16<<<1024, 256, 0, stream>>>(Wc, Wcb, HSZ * KCA);
  cvt_f32_to_bf16<<<1024, 256, 0, stream>>>(Wo, Wob, HSZ * KWF);
  init_comb<<<256, 256, 0, stream>>>(x, h0, c0, cmb[0], cfp, BATCH * HSZ);

  float* houts  = out;                                  // [512,64,1024]
  float* h_tail = out + (size_t)SEQ * BATCH * HSZ;      // h_f
  float* c_tail = h_tail + (size_t)BATCH * HSZ;         // c_f

  const dim3 grid(HSZ / 16, BATCH / 16);                // (64 N-tiles, 4 M-tiles)

  for (int t = 0; t < SEQ; ++t) {
    const int cur = t & 1, nxt = cur ^ 1;
    const bool last = (t == SEQ - 1);

    lstm_phase_a<<<grid, 32, 0, stream>>>(
        cmb[cur], cmb[nxt], cfp,
        Wwb, Wfb, Wcb, bw, bfg, bc, last ? c_tail : nullptr);

    lstm_phase_b<<<grid, 32, 0, stream>>>(
        cmb[cur], cmb[nxt], cfp,
        Wob, bo,
        last ? nullptr : x + (size_t)(t + 1) * BATCH * ISZ,
        houts + (size_t)t * BATCH * HSZ,
        last ? h_tail : nullptr);
  }
}
